// RelativeMultiheadAttention_5119601017480
// MI455X (gfx1250) — compile-verified, matrix operand order checked by reading, not run on hardware
//
#include <hip/hip_runtime.h>
#include <hip/hip_bf16.h>
#include <stdint.h>

// ---------------- problem constants ----------------
#define QLEN   1024
#define RLEN   2048
#define BATCH  2
#define DMODEL 1024
#define NHEAD  16
#define DHEAD  64
#define ATTN_SCALE 0.125f      // 1/sqrt(64)
#define NEG_BIG 1.0e30f

typedef __attribute__((ext_vector_type(16))) __bf16 v16bf;
typedef __attribute__((ext_vector_type(8)))  float  v8f;

__device__ __forceinline__ unsigned short f32_to_bf16(float f) {
    unsigned int u = __builtin_bit_cast(unsigned int, f);
    u += 0x7fffu + ((u >> 16) & 1u);          // round-to-nearest-even
    return (unsigned short)(u >> 16);
}
__device__ __forceinline__ __bf16 us_to_bf(unsigned short s) {
    return __builtin_bit_cast(__bf16, s);
}
// A/B fragment K index for 16-bit WMMA (16x32), per ISA 7.12.2
__device__ __forceinline__ int kmap(int e, int hi) {
    return ((e >> 3) << 4) + (hi << 3) + (e & 7);
}
__device__ __forceinline__ v8f wmma_bf16(v16bf a, v16bf b, v8f c) {
    return __builtin_amdgcn_wmma_f32_16x16x32_bf16(false, a, false, b, (short)0, c,
                                                   false, false);
}
// gfx1250 async DMA: global -> LDS, 16B per lane, tracked by ASYNCcnt
__device__ __forceinline__ void async_b128(void* lds, const void* gsrc) {
    unsigned loff = (unsigned)(uintptr_t)lds;   // generic LDS addr low 32b = LDS offset
    asm volatile("global_load_async_to_lds_b128 %0, %1, off"
                 :: "v"(loff), "v"(gsrc) : "memory");
}
__device__ __forceinline__ void wait_async0() {
    asm volatile("s_wait_asynccnt 0x0" ::: "memory");
}
__device__ __forceinline__ void wait_async8() {
    asm volatile("s_wait_asynccnt 0x8" ::: "memory");
}

// ---------------- pack kernels ----------------
__global__ __launch_bounds__(256) void pack_bf16(const float* __restrict__ src,
                                                 unsigned short* __restrict__ dst, int n) {
    int i = blockIdx.x * 256 + threadIdx.x;
    if (i < n) dst[i] = f32_to_bf16(src[i]);
}
// kernel [k=DMODEL][c=n*64+d]  ->  Wt[c][k]   (transposed, K contiguous per row)
__global__ __launch_bounds__(256) void pack_wt(const float* __restrict__ src,
                                               unsigned short* __restrict__ dst) {
    int idx = blockIdx.x * 256 + threadIdx.x;     // idx = c*1024 + k
    int c = idx >> 10, k = idx & 1023;
    dst[idx] = f32_to_bf16(src[k * DMODEL + c]);
}

// ---------------- bf16 WMMA GEMM: C = A(MxK) * Bt(NxK)^T ----------------
// A row-major [M][K], Bt row-major [N][K] (transposed weights).
// mode 0: C f32 [M][N]
// mode 1: C f32 head  [b][n][i][d]   (row = i*BATCH+b, col = n*64+d)
// mode 2: C bf16 head [b][n][i][d]
// mode 3: C bf16 head transposed [b][n][d][i]
__global__ __launch_bounds__(128)
void gemm_bf16_wmma(const unsigned short* __restrict__ A,
                    const unsigned short* __restrict__ Bt,
                    void* __restrict__ C,
                    int M, int N, int K, int mode, int S) {
    __shared__ __attribute__((aligned(16))) unsigned short sA[2][64 * 64];
    __shared__ __attribute__((aligned(16))) unsigned short sB[2][64 * 64];
    const int tid  = threadIdx.x;
    const int wave = tid >> 5, lane = tid & 31;
    const int lo   = lane & 15, hi = lane >> 4;
    const int m0   = blockIdx.y * 64, n0 = blockIdx.x * 64;

    v8f acc[4];
#pragma unroll
    for (int mt = 0; mt < 4; ++mt) acc[mt] = v8f{};

    // issue one 64x64 A tile + 64x64 B tile as 8 async-DMA instructions
    auto issue = [&](int buf, int kt) {
#pragma unroll
        for (int idx = tid; idx < 512; idx += 128)
            async_b128(&sA[buf][idx * 8],
                       A + (size_t)(m0 + (idx >> 3)) * K + kt + (idx & 7) * 8);
#pragma unroll
        for (int idx = tid; idx < 512; idx += 128)
            async_b128(&sB[buf][idx * 8],
                       Bt + (size_t)(n0 + (idx >> 3)) * K + kt + (idx & 7) * 8);
    };

    issue(0, 0);
    int p = 0;
    for (int kt = 0; kt < K; kt += 64) {
        const bool more = (kt + 64) < K;
        if (more) issue(1 - p, kt + 64);    // DMA next tile while computing this one
        if (more) wait_async8();            // retire previous tile's 8 ops (in-order)
        else      wait_async0();
        __syncthreads();

#pragma unroll
        for (int ks = 0; ks < 64; ks += 32) {
            v16bf bf;
#pragma unroll
            for (int e = 0; e < 16; ++e)
                bf[e] = us_to_bf(sB[p][(size_t)(wave * 16 + lo) * 64 + ks + kmap(e, hi)]);
#pragma unroll
            for (int mt = 0; mt < 4; ++mt) {
                v16bf af;
#pragma unroll
                for (int e = 0; e < 16; ++e)
                    af[e] = us_to_bf(sA[p][(size_t)(mt * 16 + lo) * 64 + ks + kmap(e, hi)]);
                acc[mt] = wmma_bf16(af, bf, acc[mt]);
            }
        }
        __syncthreads();                    // all waves done reading buf p
        p ^= 1;
    }

#pragma unroll
    for (int mt = 0; mt < 4; ++mt) {
#pragma unroll
        for (int g = 0; g < 8; ++g) {
            int row = m0 + mt * 16 + g + 8 * hi;
            int col = n0 + wave * 16 + lo;
            float v = acc[mt][g];
            if (mode == 0) {
                ((float*)C)[(size_t)row * N + col] = v;
            } else {
                int i = row >> 1, b = row & 1;        // BATCH == 2
                int n = col >> 6, d = col & 63;
                if (mode == 1) {
                    ((float*)C)[(((size_t)(b * NHEAD + n) * S + i) * DHEAD) + d] = v;
                } else if (mode == 2) {
                    ((unsigned short*)C)[(((size_t)(b * NHEAD + n) * S + i) * DHEAD) + d] =
                        f32_to_bf16(v);
                } else {
                    ((unsigned short*)C)[(((size_t)(b * NHEAD + n) * DHEAD + d) * S) + i] =
                        f32_to_bf16(v);
                }
            }
        }
    }
}

// ---------------- bias add (bf16) + segment-embedding dot (ef) ----------------
__global__ __launch_bounds__(32)
void bias_ef_kernel(const float* __restrict__ qhead,
                    const float* __restrict__ rwb, const float* __restrict__ rrb,
                    const float* __restrict__ rsb, const float* __restrict__ semb,
                    unsigned short* __restrict__ qw, unsigned short* __restrict__ qr,
                    float* __restrict__ ef) {
    const int i = blockIdx.x, nh = blockIdx.y, b = blockIdx.z, lane = threadIdx.x;
    const size_t base = (((size_t)(b * NHEAD + nh)) * QLEN + i) * DHEAD;
    float e0 = 0.f, e1 = 0.f;
#pragma unroll
    for (int t = 0; t < 2; ++t) {
        int d = lane + t * 32;
        float q = qhead[base + d];
        qw[base + d] = f32_to_bf16(q + rwb[nh * DHEAD + d]);
        qr[base + d] = f32_to_bf16(q + rrb[nh * DHEAD + d]);
        float qs = q + rsb[nh * DHEAD + d];
        e0 += qs * semb[nh * DHEAD + d];
        e1 += qs * semb[NHEAD * DHEAD + nh * DHEAD + d];
    }
#pragma unroll
    for (int off = 16; off >= 1; off >>= 1) {
        e0 += __shfl_xor(e0, off, 32);
        e1 += __shfl_xor(e1, off, 32);
    }
    if (lane == 0) {
        size_t o = (((size_t)(b * NHEAD + nh)) * QLEN + i) * 2;
        ef[o + 0] = e0;
        ef[o + 1] = e1;
    }
}

// ---------------- fused flash attention with rel-shift, one wave / 16 queries ----
// kh: [b][n][key][d],  vt: [b][n][d][key] (transposed),  kr: [b][n][pos][d]
__global__ __launch_bounds__(32)
void flash_attn_wmma(const unsigned short* __restrict__ qw,
                     const unsigned short* __restrict__ qr,
                     const unsigned short* __restrict__ kh,
                     const unsigned short* __restrict__ vt,
                     const unsigned short* __restrict__ kr,
                     const float* __restrict__ ef,
                     const unsigned char* __restrict__ seg,
                     const float* __restrict__ amask,
                     unsigned short* __restrict__ av) {
    __shared__ __attribute__((aligned(16))) unsigned short sK[32 * 64];   // [key][d]
    __shared__ __attribute__((aligned(16))) unsigned short sVt[64 * 32];  // [d][key]
    __shared__ __attribute__((aligned(16))) unsigned short sKR[48 * 64];  // [pos][d]
    __shared__ unsigned short sP[16 * 32];
    __shared__ float sBD[16 * 48];
    __shared__ float sEF[16 * 2];

    const int lane = threadIdx.x;
    const int lo = lane & 15, hi = lane >> 4;
    const int i0 = blockIdx.x * 16;
    const int nh = blockIdx.y;
    const int b  = blockIdx.z;
    const size_t headQ = (size_t)(b * NHEAD + nh) * QLEN;
    const size_t headR = (size_t)(b * NHEAD + nh) * RLEN;

    // persistent query fragments (q + r_w_bias), (q + r_r_bias), K-halves 0..31 / 32..63
    v16bf aW0, aW1, aR0, aR1;
    {
        const unsigned short* qb = qw + (headQ + i0) * DHEAD;
        const unsigned short* rb = qr + (headQ + i0) * DHEAD;
#pragma unroll
        for (int e = 0; e < 16; ++e) {
            int k = kmap(e, hi);
            aW0[e] = us_to_bf(qb[(size_t)lo * DHEAD + k]);
            aW1[e] = us_to_bf(qb[(size_t)lo * DHEAD + 32 + k]);
            aR0[e] = us_to_bf(rb[(size_t)lo * DHEAD + k]);
            aR1[e] = us_to_bf(rb[(size_t)lo * DHEAD + 32 + k]);
        }
    }
    if (lane < 16) {
        sEF[lane * 2 + 0] = ef[(headQ + i0 + lane) * 2 + 0];
        sEF[lane * 2 + 1] = ef[(headQ + i0 + lane) * 2 + 1];
    }

    float m_run[8], l_run[8];
    v8f O[4];
#pragma unroll
    for (int g = 0; g < 8; ++g) { m_run[g] = -3.0e38f; l_run[g] = 0.f; }
#pragma unroll
    for (int t = 0; t < 4; ++t) O[t] = v8f{};

    for (int jt = 0; jt < QLEN / 32; ++jt) {
        const int j0 = jt * 32;
        const int base = j0 - i0 + QLEN - 15;   // kr window start; bd[i,j]=qr_i . kr[j-i+Q]
        const unsigned short* kb = kh + (headQ + j0) * DHEAD;
        const unsigned short* vb = vt + (size_t)(b * NHEAD + nh) * DHEAD * QLEN + j0;
        const unsigned short* rb = kr + (headR + base) * DHEAD;
        // async DMA tiles into LDS (28 ops), then one asynccnt wait
#pragma unroll
        for (int idx = lane; idx < 256; idx += 32) {
            async_b128(&sK[idx * 8],  kb + (size_t)(idx >> 3) * DHEAD + (idx & 7) * 8);
            async_b128(&sVt[idx * 8], vb + (size_t)(idx >> 2) * QLEN + (idx & 3) * 8);
        }
#pragma unroll
        for (int idx = lane; idx < 384; idx += 32)
            async_b128(&sKR[idx * 8], rb + (size_t)(idx >> 3) * DHEAD + (idx & 7) * 8);
        wait_async0();
        __syncthreads();

        // content scores: two 16x16 N-tiles, K=64 -> 4 WMMAs
        v8f accS[2];
#pragma unroll
        for (int t = 0; t < 2; ++t) {
            v16bf b0, b1;
#pragma unroll
            for (int e = 0; e < 16; ++e) {
                int k = kmap(e, hi);
                b0[e] = us_to_bf(sK[(size_t)(t * 16 + lo) * DHEAD + k]);
                b1[e] = us_to_bf(sK[(size_t)(t * 16 + lo) * DHEAD + 32 + k]);
            }
            v8f a = v8f{};
            a = wmma_bf16(aW0, b0, a);
            a = wmma_bf16(aW1, b1, a);
            accS[t] = a;
        }
        // rel-position panel P[mm, c] over 47-row kr window: 3 N-tiles -> 6 WMMAs
#pragma unroll
        for (int t = 0; t < 3; ++t) {
            v16bf b0, b1;
#pragma unroll
            for (int e = 0; e < 16; ++e) {
                int k = kmap(e, hi);
                b0[e] = us_to_bf(sKR[(size_t)(t * 16 + lo) * DHEAD + k]);
                b1[e] = us_to_bf(sKR[(size_t)(t * 16 + lo) * DHEAD + 32 + k]);
            }
            v8f a = v8f{};
            a = wmma_bf16(aR0, b0, a);
            a = wmma_bf16(aR1, b1, a);
#pragma unroll
            for (int g = 0; g < 8; ++g)
                sBD[(g + 8 * hi) * 48 + t * 16 + lo] = a[g];
        }
        __syncthreads();

        // combine ac + shifted bd + ef, mask, scale
        float sv[2][8];
#pragma unroll
        for (int t = 0; t < 2; ++t) {
#pragma unroll
            for (int g = 0; g < 8; ++g) {
                int mm = g + 8 * hi;
                int jj = t * 16 + lo;
                int i = i0 + mm, j = j0 + jj;
                size_t ij = ((size_t)i * QLEN + j) * BATCH + b;
                float s = accS[t][g] + sBD[mm * 48 + jj + 15 - mm] +
                          (seg[ij] ? sEF[mm * 2 + 1] : sEF[mm * 2 + 0]);
                sv[t][g] = s * ATTN_SCALE - NEG_BIG * amask[ij];
            }
        }
        // online softmax (rows live in 16-lane halves)
#pragma unroll
        for (int g = 0; g < 8; ++g) {
            float mx = fmaxf(sv[0][g], sv[1][g]);
#pragma unroll
            for (int off = 8; off >= 1; off >>= 1)
                mx = fmaxf(mx, __shfl_xor(mx, off, 16));
            float mnew  = fmaxf(m_run[g], mx);
            float alpha = __expf(m_run[g] - mnew);
            float p0 = __expf(sv[0][g] - mnew);
            float p1 = __expf(sv[1][g] - mnew);
            float rs = p0 + p1;
#pragma unroll
            for (int off = 8; off >= 1; off >>= 1)
                rs += __shfl_xor(rs, off, 16);
            l_run[g] = l_run[g] * alpha + rs;
            m_run[g] = mnew;
            int mm = g + 8 * hi;
            sP[mm * 32 + lo]      = f32_to_bf16(p0);
            sP[mm * 32 + 16 + lo] = f32_to_bf16(p1);
#pragma unroll
            for (int t = 0; t < 4; ++t) O[t][g] *= alpha;
        }
        __syncthreads();

        // O += P(16x32) x V(32x64): 4 WMMAs (V staged transposed -> contiguous reads)
        v16bf pf;
#pragma unroll
        for (int e = 0; e < 16; ++e) pf[e] = us_to_bf(sP[lo * 32 + kmap(e, hi)]);
#pragma unroll
        for (int t = 0; t < 4; ++t) {
            v16bf vf;
#pragma unroll
            for (int e = 0; e < 16; ++e)
                vf[e] = us_to_bf(sVt[(size_t)(t * 16 + lo) * 32 + kmap(e, hi)]);
            O[t] = wmma_bf16(pf, vf, O[t]);
        }
        __syncthreads();
    }

    // normalize and store attn_vec as bf16 A-matrix for the output projection
#pragma unroll
    for (int g = 0; g < 8; ++g) {
        int mm = g + 8 * hi;
        float inv = 1.0f / l_run[g];
        size_t rowbase = ((size_t)(i0 + mm) * BATCH + b) * DMODEL + nh * DHEAD;
#pragma unroll
        for (int t = 0; t < 4; ++t)
            av[rowbase + t * 16 + lo] = f32_to_bf16(O[t][g] * inv);
    }
}

// ---------------- residual + LayerNorm ----------------
__global__ __launch_bounds__(256)
void residual_ln_kernel(const float* __restrict__ x, const float* __restrict__ h,
                        const float* __restrict__ gamma, const float* __restrict__ beta,
                        float* __restrict__ out) {
    __shared__ float red[256];
    const int row = blockIdx.x, tid = threadIdx.x;
    float v[4];
    float s = 0.f;
#pragma unroll
    for (int t = 0; t < 4; ++t) {
        int c = tid + t * 256;
        v[t] = x[(size_t)row * DMODEL + c] + h[(size_t)row * DMODEL + c];
        s += v[t];
    }
    red[tid] = s;
    __syncthreads();
    for (int off = 128; off > 0; off >>= 1) {
        if (tid < off) red[tid] += red[tid + off];
        __syncthreads();
    }
    float mu = red[0] * (1.0f / DMODEL);
    __syncthreads();
    float s2 = 0.f;
#pragma unroll
    for (int t = 0; t < 4; ++t) { float d = v[t] - mu; s2 += d * d; }
    red[tid] = s2;
    __syncthreads();
    for (int off = 128; off > 0; off >>= 1) {
        if (tid < off) red[tid] += red[tid + off];
        __syncthreads();
    }
    float inv = rsqrtf(red[0] * (1.0f / DMODEL) + 1e-12f);
#pragma unroll
    for (int t = 0; t < 4; ++t) {
        int c = tid + t * 256;
        out[(size_t)row * DMODEL + c] = (v[t] - mu) * inv * gamma[c] + beta[c];
    }
}

// ---------------- host ----------------
extern "C" void kernel_launch(void* const* d_in, const int* in_sizes, int n_in,
                              void* d_out, int out_size, void* d_ws, size_t ws_size,
                              hipStream_t stream) {
    (void)in_sizes; (void)n_in; (void)out_size; (void)ws_size;
    const float* h     = (const float*)d_in[0];
    const float* r     = (const float*)d_in[1];
    const unsigned char* seg = (const unsigned char*)d_in[2];
    const float* amask = (const float*)d_in[3];
    const float* qk    = (const float*)d_in[4];
    const float* kk    = (const float*)d_in[5];
    const float* vk    = (const float*)d_in[6];
    const float* rk    = (const float*)d_in[7];
    const float* ok    = (const float*)d_in[8];
    const float* rwb   = (const float*)d_in[9];
    const float* rrb   = (const float*)d_in[10];
    const float* rsb   = (const float*)d_in[11];
    const float* semb  = (const float*)d_in[12];
    const float* gamma = (const float*)d_in[13];
    const float* beta  = (const float*)d_in[14];
    float* out = (float*)d_out;

    char* w = (char*)d_ws;
    auto carve = [&](size_t bytes) {
        char* p = w;
        w += (bytes + 255) & ~(size_t)255;
        return p;
    };
    unsigned short* h_bf  = (unsigned short*)carve((size_t)QLEN * BATCH * DMODEL * 2);
    unsigned short* r_bf  = (unsigned short*)carve((size_t)RLEN * BATCH * DMODEL * 2);
    unsigned short* Wq    = (unsigned short*)carve((size_t)DMODEL * DMODEL * 2);
    unsigned short* Wk    = (unsigned short*)carve((size_t)DMODEL * DMODEL * 2);
    unsigned short* Wv    = (unsigned short*)carve((size_t)DMODEL * DMODEL * 2);
    unsigned short* Wr    = (unsigned short*)carve((size_t)DMODEL * DMODEL * 2);
    unsigned short* Wo    = (unsigned short*)carve((size_t)DMODEL * DMODEL * 2);
    float*          qhead = (float*)carve((size_t)BATCH * NHEAD * QLEN * DHEAD * 4);
    unsigned short* qw_bf = (unsigned short*)carve((size_t)BATCH * NHEAD * QLEN * DHEAD * 2);
    unsigned short* qr_bf = (unsigned short*)carve((size_t)BATCH * NHEAD * QLEN * DHEAD * 2);
    unsigned short* k_bf  = (unsigned short*)carve((size_t)BATCH * NHEAD * QLEN * DHEAD * 2);
    unsigned short* v_bf  = (unsigned short*)carve((size_t)BATCH * NHEAD * QLEN * DHEAD * 2);
    unsigned short* kr_bf = (unsigned short*)carve((size_t)BATCH * NHEAD * RLEN * DHEAD * 2);
    float*          efbuf = (float*)carve((size_t)BATCH * NHEAD * QLEN * 2 * 4);
    unsigned short* av_bf = (unsigned short*)carve((size_t)QLEN * BATCH * DMODEL * 2);
    float*          tmp   = (float*)carve((size_t)QLEN * BATCH * DMODEL * 4);

    int n;
    n = QLEN * BATCH * DMODEL; pack_bf16<<<n / 256, 256, 0, stream>>>(h, h_bf, n);
    n = RLEN * BATCH * DMODEL; pack_bf16<<<n / 256, 256, 0, stream>>>(r, r_bf, n);
    n = DMODEL * NHEAD * DHEAD;
    pack_wt<<<n / 256, 256, 0, stream>>>(qk, Wq);
    pack_wt<<<n / 256, 256, 0, stream>>>(kk, Wk);
    pack_wt<<<n / 256, 256, 0, stream>>>(vk, Wv);
    pack_wt<<<n / 256, 256, 0, stream>>>(rk, Wr);
    pack_bf16<<<n / 256, 256, 0, stream>>>(ok, Wo, n);   // o_kernel [h'][nd] is already [N][K]

    dim3 gb(128);
    gemm_bf16_wmma<<<dim3(16, 32), gb, 0, stream>>>(h_bf, Wq, qhead, 2048, 1024, 1024, 1, QLEN);
    gemm_bf16_wmma<<<dim3(16, 32), gb, 0, stream>>>(h_bf, Wk, k_bf, 2048, 1024, 1024, 2, QLEN);
    gemm_bf16_wmma<<<dim3(16, 32), gb, 0, stream>>>(h_bf, Wv, v_bf, 2048, 1024, 1024, 3, QLEN);
    gemm_bf16_wmma<<<dim3(16, 64), gb, 0, stream>>>(r_bf, Wr, kr_bf, 4096, 1024, 1024, 2, RLEN);

    bias_ef_kernel<<<dim3(QLEN, NHEAD, BATCH), 32, 0, stream>>>(qhead, rwb, rrb, rsb, semb,
                                                                qw_bf, qr_bf, efbuf);

    flash_attn_wmma<<<dim3(QLEN / 16, NHEAD, BATCH), 32, 0, stream>>>(
        qw_bf, qr_bf, k_bf, v_bf, kr_bf, efbuf, seg, amask, av_bf);

    gemm_bf16_wmma<<<dim3(16, 32), gb, 0, stream>>>(av_bf, Wo, tmp, 2048, 1024, 1024, 0, QLEN);

    residual_ln_kernel<<<QLEN * BATCH, 256, 0, stream>>>(tmp, h, gamma, beta, out);
}
